// TTGNN_32633161515661
// MI455X (gfx1250) — matrix operand order repaired and checked by reading, not compiled, not run on hardware
//
#include <hip/hip_runtime.h>
#include <stdint.h>

#define NN 20000
#define EE 120000
#define DD 1024
#define HH 8
#define CC 128
#define LL 2

typedef __attribute__((ext_vector_type(16))) __bf16        v16bf;
typedef __attribute__((ext_vector_type(8)))  float         v8f;
typedef __attribute__((ext_vector_type(8)))  unsigned int  v8u;

// ---------- helpers ----------
__device__ __forceinline__ unsigned short f2bf(float f) {
  unsigned u = __float_as_uint(f);
  u += 0x7fffu + ((u >> 16) & 1u);            // round-to-nearest-even
  return (unsigned short)(u >> 16);
}
__device__ __forceinline__ unsigned fenc(float f) {   // order-preserving float->u32
  unsigned u = __float_as_uint(f);
  return (u & 0x80000000u) ? ~u : (u | 0x80000000u);
}
__device__ __forceinline__ float fdec(unsigned u) {
  return __uint_as_float((u & 0x80000000u) ? (u & 0x7fffffffu) : ~u);
}

// ---------- tiny elementwise kernels ----------
__global__ void k_counts(const int* __restrict__ dst, const int* __restrict__ eattr,
                         float* __restrict__ cnt, float* __restrict__ deg) {
  int e = blockIdx.x * blockDim.x + threadIdx.x;
  if (e >= EE) return;
  int d = dst[e];
  atomicAdd(&deg[d], 1.0f);
  atomicAdd(&cnt[(size_t)d * 5 + eattr[e]], 1.0f);
}

__global__ void k_h(const float* __restrict__ x, const int* __restrict__ ntype,
                    const float* __restrict__ ntemb, float* __restrict__ h) {
  size_t t = (size_t)blockIdx.x * blockDim.x + threadIdx.x;
  int n = (int)(t >> 10), d = (int)(t & (DD - 1));
  h[t] = x[t] + ntemb[(size_t)ntype[n] * DD + d];
}

__global__ void k_cvt(const float* __restrict__ in, unsigned short* __restrict__ out, int nelem) {
  int t = blockIdx.x * blockDim.x + threadIdx.x;
  if (t < nelem) out[t] = f2bf(in[t]);
}

// Wt[n][k] = W[k][n] in bf16 (so B fragments become contiguous K-runs)
__global__ void k_wt(const float* __restrict__ W, unsigned short* __restrict__ Wt) {
  int t = blockIdx.x * blockDim.x + threadIdx.x;
  int kk = t >> 10, n = t & (DD - 1);
  Wt[(size_t)n * DD + kk] = f2bf(W[(size_t)kk * DD + n]);
}

// T[t][d] = sum_k edge_type_emb[t][k] * W_e[k][d]   (5 x 1024)
__global__ void k_T(const float* __restrict__ ete, const float* __restrict__ We,
                    float* __restrict__ T) {
  int tid = blockIdx.x * blockDim.x + threadIdx.x;   // 5*1024
  int t = tid >> 10, d = tid & (DD - 1);
  float s = 0.f;
  for (int k = 0; k < DD; ++k) s += ete[(size_t)t * DD + k] * We[(size_t)k * DD + d];
  T[tid] = s;
}

// ee_loop[n][d] = sum_t cnt[n][t] * T[t][d] / max(deg[n],1)
__global__ void k_eeloop(const float* __restrict__ cnt, const float* __restrict__ deg,
                         const float* __restrict__ T, float* __restrict__ out) {
  size_t tid = (size_t)blockIdx.x * blockDim.x + threadIdx.x;
  int n = (int)(tid >> 10), d = (int)(tid & (DD - 1));
  float inv = 1.0f / fmaxf(deg[n], 1.0f);
  float s = 0.f;
#pragma unroll
  for (int t = 0; t < 5; ++t) s += cnt[(size_t)n * 5 + t] * T[t * DD + d];
  out[tid] = s * inv;
}

// ---------- bf16 WMMA GEMM: C[M x 1024] = A[M x 1024] @ W + epilogue ----------
// A row-major bf16, Bt = W^T row-major bf16 (Bt[n][k]).
// Double-buffered LDS tiles filled with GLOBAL_LOAD_ASYNC_TO_LDS_B128 (ASYNCcnt),
// one workgroup barrier per K-step, DMA of next tile overlaps WMMAs of current.
// mode 0: C = acc + bias ;  mode 1: C = Xres + gate[0]*(acc + bias)
__global__ __launch_bounds__(256) void gemm_bf16_wmma(
    const unsigned short* __restrict__ A, const unsigned short* __restrict__ Bt,
    const float* __restrict__ bias, float* __restrict__ C,
    const float* __restrict__ Xres, const float* __restrict__ gate,
    int M, int mode) {
  __shared__ __align__(16) unsigned short As[2][128][32];
  __shared__ __align__(16) unsigned short Bs[2][128][32];

  const int tid = threadIdx.x;
  const int wave = tid >> 5, lane = tid & 31;
  const int wm = wave & 3, wn = wave >> 2;           // 4x2 wave grid
  const int mBase = blockIdx.x * 128;
  const int nBase = blockIdx.y * 128;
  const int lr = lane & 15, lh = lane >> 4;

  const v8f zf = {0.f, 0.f, 0.f, 0.f, 0.f, 0.f, 0.f, 0.f};
  v8f acc[2][4];
#pragma unroll
  for (int i = 0; i < 2; ++i)
#pragma unroll
    for (int j = 0; j < 4; ++j) acc[i][j] = zf;

  const int ldRow = tid >> 1;            // 0..127
  const int ldSeg = (tid & 1) * 16;      // 0 or 16 (shorts)
  int ar = mBase + ldRow;
  if (ar >= M) ar = M - 1;  // clamp: garbage A-rows only affect store-guarded out rows
  const unsigned short* gA = A + (size_t)ar * DD + ldSeg;
  const unsigned short* gB = Bt + (size_t)(nBase + ldRow) * DD + ldSeg;

  // issue 4 async DMA loads (32B of A tile + 32B of B tile) for stage pb, K-offset k0.
  // INST_OFFSET is added to BOTH the LDS and global address (ISA 10.7/15.18), so one
  // LDS-address VGPR + one 64-bit global address covers both 16B chunks.
  auto stage = [&](int pb, int k0) {
    unsigned aL = (unsigned)(size_t)&As[pb][ldRow][ldSeg];   // low 32b = LDS byte offset
    unsigned bL = (unsigned)(size_t)&Bs[pb][ldRow][ldSeg];
    unsigned long long ga = (unsigned long long)(size_t)(gA + k0);
    unsigned long long gb = (unsigned long long)(size_t)(gB + k0);
    asm volatile(
        "global_load_async_to_lds_b128 %0, %2, off\n\t"
        "global_load_async_to_lds_b128 %0, %2, off offset:16\n\t"
        "global_load_async_to_lds_b128 %1, %3, off\n\t"
        "global_load_async_to_lds_b128 %1, %3, off offset:16"
        :: "v"(aL), "v"(bL), "v"(ga), "v"(gb) : "memory");
  };

  // prologue: fill buffer 0
  stage(0, 0);
  asm volatile("s_wait_asynccnt 0x0" ::: "memory");
  __syncthreads();

  int p = 0;
  for (int k0 = 0; k0 < DD; k0 += 32) {
    if (k0 + 32 < DD) stage(p ^ 1, k0 + 32);       // DMA next tile, overlaps compute

    // --- build fragments (ISA 16-bit A/B VGPR layouts, wave32) ---
    v16bf afr[2], bfr[4];
#pragma unroll
    for (int i = 0; i < 2; ++i) {
      int r = wm * 32 + i * 16 + lr;
      uint4 lo = *(const uint4*)&As[p][r][lh * 8];        // K = lh*8 .. +7
      uint4 hi = *(const uint4*)&As[p][r][16 + lh * 8];   // K = 16+lh*8 .. +7
      v8u u = {lo.x, lo.y, lo.z, lo.w, hi.x, hi.y, hi.z, hi.w};
      afr[i] = __builtin_bit_cast(v16bf, u);
    }
#pragma unroll
    for (int j = 0; j < 4; ++j) {
      int c = wn * 64 + j * 16 + lr;
      uint4 lo = *(const uint4*)&Bs[p][c][lh * 8];        // K = lh*8 .. +7
      uint4 hi = *(const uint4*)&Bs[p][c][16 + lh * 8];   // K = 16+lh*8 .. +7
      v8u u = {lo.x, lo.y, lo.z, lo.w, hi.x, hi.y, hi.z, hi.w};
      bfr[j] = __builtin_bit_cast(v16bf, u);
    }
#pragma unroll
    for (int i = 0; i < 2; ++i)
#pragma unroll
      for (int j = 0; j < 4; ++j)
        acc[i][j] = __builtin_amdgcn_wmma_f32_16x16x32_bf16(
            false, afr[i], false, bfr[j], (short)0, acc[i][j], false, false);

    // next-stage DMA done (it had the whole compute phase), then one barrier:
    // all waves finished reading buf p and writing buf p^1.
    asm volatile("s_wait_asynccnt 0x0" ::: "memory");
    __syncthreads();
    p ^= 1;
  }

  // --- epilogue (32-bit C/D layout: lane->N, vgpr->M, +8 for hi lanes) ---
  const float g = (mode == 1) ? gate[0] : 0.f;
#pragma unroll
  for (int i = 0; i < 2; ++i)
#pragma unroll
    for (int j = 0; j < 4; ++j) {
      int col = nBase + wn * 64 + j * 16 + lr;
      float bv = bias[col];
#pragma unroll
      for (int r = 0; r < 8; ++r) {
        int row = mBase + wm * 32 + i * 16 + r + (lh ? 8 : 0);
        if (row < M) {
          size_t off = (size_t)row * DD + col;
          float v = acc[i][j][r] + bv;
          C[off] = (mode == 1) ? (Xres[off] + g * v) : v;
        }
      }
    }
}

// ---------- attention logits: one wave per edge ----------
__global__ __launch_bounds__(256) void k_logits(
    const float* __restrict__ xl, const float* __restrict__ xr,
    const float* __restrict__ T, const float* __restrict__ eeloop,
    const int* __restrict__ src, const int* __restrict__ dst,
    const int* __restrict__ eattr, const float* __restrict__ att,
    float* __restrict__ lg, unsigned* __restrict__ mx) {
  int wid = (blockIdx.x * blockDim.x + threadIdx.x) >> 5;
  int lane = threadIdx.x & 31;
  if (wid >= EE + NN) return;
  int s, d; const float* er;
  if (wid < EE) { s = src[wid]; d = dst[wid]; er = T + (size_t)eattr[wid] * DD; }
  else          { s = d = wid - EE;           er = eeloop + (size_t)s * DD;     }
  const float* pl = xl + (size_t)s * DD;
  const float* pr = xr + (size_t)d * DD;
  float acc[HH];
#pragma unroll
  for (int hh = 0; hh < HH; ++hh) acc[hh] = 0.f;
#pragma unroll
  for (int hh = 0; hh < HH; ++hh)
#pragma unroll
    for (int k = 0; k < 4; ++k) {
      int idx = lane + (hh * 4 + k) * 32;          // coalesced across lanes
      float v = pl[idx] + pr[idx] + er[idx];
      v = v > 0.f ? v : 0.2f * v;                  // leaky_relu(0.2)
      acc[hh] += v * att[hh * CC + (idx & (CC - 1))];
    }
#pragma unroll
  for (int hh = 0; hh < HH; ++hh)
#pragma unroll
    for (int o = 16; o > 0; o >>= 1) acc[hh] += __shfl_xor(acc[hh], o, 32);
  if (lane == 0) {
#pragma unroll
    for (int hh = 0; hh < HH; ++hh) {
      lg[(size_t)wid * HH + hh] = acc[hh];
      atomicMax(&mx[(size_t)d * HH + hh], fenc(acc[hh]));
    }
  }
}

__global__ void k_exp(float* __restrict__ lg, const unsigned* __restrict__ mx,
                      float* __restrict__ den, const int* __restrict__ dst) {
  int tid = blockIdx.x * blockDim.x + threadIdx.x;
  if (tid >= (EE + NN) * HH) return;
  int e = tid >> 3, hh = tid & 7;
  int d = (e < EE) ? dst[e] : (e - EE);
  float v = __expf(lg[tid] - fdec(mx[(size_t)d * HH + hh]));
  lg[tid] = v;
  atomicAdd(&den[(size_t)d * HH + hh], v);
}

// ---------- aggregation: one wave per edge, scatter-add ----------
__global__ __launch_bounds__(256) void k_agg(
    const float* __restrict__ xl, const float* __restrict__ ex,
    const float* __restrict__ den, const int* __restrict__ src,
    const int* __restrict__ dst, float* __restrict__ agg) {
  int wid = (blockIdx.x * blockDim.x + threadIdx.x) >> 5;
  int lane = threadIdx.x & 31;
  if (wid >= EE + NN) return;
  int s, d;
  if (wid < EE) { s = src[wid]; d = dst[wid]; } else { s = d = wid - EE; }
  const float* pl = xl + (size_t)s * DD;
  float* pa = agg + (size_t)d * DD;
#pragma unroll
  for (int hh = 0; hh < HH; ++hh) {
    float a = ex[(size_t)wid * HH + hh] / den[(size_t)d * HH + hh];
#pragma unroll
    for (int k = 0; k < 4; ++k) {
      int idx = lane + (hh * 4 + k) * 32;
      atomicAdd(&pa[idx], pl[idx] * a);
    }
  }
}

// ---------- bias + ELU + residual + LayerNorm (one block per node) ----------
__global__ __launch_bounds__(256) void k_post(const float* __restrict__ agg,
                                              const float* __restrict__ bias,
                                              const float* __restrict__ gamma,
                                              const float* __restrict__ beta,
                                              float* __restrict__ h) {
  int n = blockIdx.x, t = threadIdx.x;
  const float* ar = agg + (size_t)n * DD;
  float* hr = h + (size_t)n * DD;
  float y[4], s = 0.f, s2 = 0.f;
#pragma unroll
  for (int k = 0; k < 4; ++k) {
    int i = t + 256 * k;
    float v = ar[i] + bias[i];
    v = v > 0.f ? v : expm1f(v);                 // elu
    float yy = v + hr[i];                        // residual
    y[k] = yy; s += yy; s2 += yy * yy;
  }
#pragma unroll
  for (int o = 16; o > 0; o >>= 1) { s += __shfl_xor(s, o, 32); s2 += __shfl_xor(s2, o, 32); }
  __shared__ float rs[8], rs2[8];
  int w = t >> 5, lane = t & 31;
  if (lane == 0) { rs[w] = s; rs2[w] = s2; }
  __syncthreads();
  if (t == 0) {
    float a = 0.f, b = 0.f;
    for (int i = 0; i < 8; ++i) { a += rs[i]; b += rs2[i]; }
    rs[0] = a; rs2[0] = b;
  }
  __syncthreads();
  float mean = rs[0] * (1.f / DD);
  float var = rs2[0] * (1.f / DD) - mean * mean;
  float inv = rsqrtf(var + 1e-5f);
#pragma unroll
  for (int k = 0; k < 4; ++k) {
    int i = t + 256 * k;
    hr[i] = (y[k] - mean) * inv * gamma[i] + beta[i];
  }
}

// ---------- launcher ----------
extern "C" void kernel_launch(void* const* d_in, const int* in_sizes, int n_in,
                              void* d_out, int out_size, void* d_ws, size_t ws_size,
                              hipStream_t stream) {
  (void)in_sizes; (void)n_in; (void)out_size; (void)ws_size;
  const float* x     = (const float*)d_in[0];
  const int*   eidx  = (const int*)d_in[1];
  const int*   eattr = (const int*)d_in[2];
  const int*   ntype = (const int*)d_in[3];
  const float* ntemb = (const float*)d_in[4];
  const float* etemb = (const float*)d_in[5];
  const float* W_l   = (const float*)d_in[6];
  const float* b_l   = (const float*)d_in[7];
  const float* W_r   = (const float*)d_in[8];
  const float* b_r   = (const float*)d_in[9];
  const float* W_e   = (const float*)d_in[10];
  const float* att   = (const float*)d_in[11];
  const float* bias  = (const float*)d_in[12];
  const float* lng   = (const float*)d_in[13];
  const float* lnb   = (const float*)d_in[14];
  const float* W_out = (const float*)d_in[15];
  const float* b_out = (const float*)d_in[16];
  const float* gate  = (const float*)d_in[17];
  const int* src = eidx;
  const int* dst = eidx + EE;

  char* ws = (char*)d_ws;
  size_t off = 0;
  auto take = [&](size_t b) -> char* {
    char* p = ws + off;
    off = (off + b + 255) & ~(size_t)255;
    return p;
  };
  float*          h    = (float*)take((size_t)NN * DD * 4);
  unsigned short* hb   = (unsigned short*)take((size_t)NN * DD * 2);
  float*          xl   = (float*)take((size_t)NN * DD * 4);
  float*          xr   = (float*)take((size_t)NN * DD * 4);
  float*          ebuf = (float*)take((size_t)NN * DD * 4);   // ee_loop, reused as agg
  unsigned short* wt   = (unsigned short*)take((size_t)DD * DD * 2);
  float*          T    = (float*)take((size_t)5 * DD * 4);
  float*          cnt  = (float*)take((size_t)NN * 5 * 4);
  float*          deg  = (float*)take((size_t)NN * 4);
  float*          lg   = (float*)take((size_t)(EE + NN) * HH * 4);
  unsigned*       mx   = (unsigned*)take((size_t)NN * HH * 4);
  float*          den  = (float*)take((size_t)NN * HH * 4);

  hipMemsetAsync(cnt, 0, (size_t)NN * 5 * 4, stream);
  hipMemsetAsync(deg, 0, (size_t)NN * 4, stream);
  k_counts<<<(EE + 255) / 256, 256, 0, stream>>>(dst, eattr, cnt, deg);
  k_h<<<(NN * DD) / 256, 256, 0, stream>>>(x, ntype, ntemb, h);

  dim3 gg((NN + 127) / 128, DD / 128);
  for (int l = 0; l < LL; ++l) {
    k_cvt<<<(NN * DD) / 256, 256, 0, stream>>>(h, hb, NN * DD);
    k_wt<<<(DD * DD) / 256, 256, 0, stream>>>(W_l + (size_t)l * DD * DD, wt);
    gemm_bf16_wmma<<<gg, 256, 0, stream>>>(hb, wt, b_l + l * DD, xl, nullptr, nullptr, NN, 0);
    k_wt<<<(DD * DD) / 256, 256, 0, stream>>>(W_r + (size_t)l * DD * DD, wt);
    gemm_bf16_wmma<<<gg, 256, 0, stream>>>(hb, wt, b_r + l * DD, xr, nullptr, nullptr, NN, 0);
    k_T<<<(5 * DD) / 256, 256, 0, stream>>>(etemb, W_e + (size_t)l * DD * DD, T);
    k_eeloop<<<(NN * DD) / 256, 256, 0, stream>>>(cnt, deg, T, ebuf);
    hipMemsetAsync(mx, 0, (size_t)NN * HH * 4, stream);
    hipMemsetAsync(den, 0, (size_t)NN * HH * 4, stream);
    k_logits<<<((EE + NN) + 7) / 8, 256, 0, stream>>>(xl, xr, T, ebuf, src, dst, eattr,
                                                      att + (size_t)l * HH * CC, lg, mx);
    k_exp<<<((EE + NN) * HH + 255) / 256, 256, 0, stream>>>(lg, mx, den, dst);
    hipMemsetAsync(ebuf, 0, (size_t)NN * DD * 4, stream);
    k_agg<<<((EE + NN) + 7) / 8, 256, 0, stream>>>(xl, lg, den, src, dst, ebuf);
    k_post<<<NN, 256, 0, stream>>>(ebuf, bias + l * DD, lng + l * DD, lnb + l * DD, h);
  }
  // final: out = x + gate * (h @ W_out + b_out)
  k_cvt<<<(NN * DD) / 256, 256, 0, stream>>>(h, hb, NN * DD);
  k_wt<<<(DD * DD) / 256, 256, 0, stream>>>(W_out, wt);
  gemm_bf16_wmma<<<gg, 256, 0, stream>>>(hb, wt, b_out, (float*)d_out, x, gate, NN, 1);
}
